// MoERegressor_34230889349393
// MI455X (gfx1250) — compile-verified, matrix operand order checked by reading, not run on hardware
//
#include <hip/hip_runtime.h>

typedef __attribute__((ext_vector_type(16))) _Float16 v16h;
typedef __attribute__((ext_vector_type(8)))  _Float16 v8h;
typedef __attribute__((ext_vector_type(8)))  float    v8f;

#define WAVES_PER_BLOCK 4
#define TOKENS_PER_WAVE 16
#define TOKENS_PER_BLOCK (WAVES_PER_BLOCK * TOKENS_PER_WAVE)

// ---- packed weight layout in d_ws (offsets in _Float16 elements) ----
// Each fragment = 512 halves: lane-major (32) x 16 halves, matching the
// 16-bit B-matrix (KxN) VGPR layout of V_WMMA_F32_16X16X32_F16.
enum : int {
  OFF_WP  = 0,        // 153x153 -> 160x160 : nk=5, nn=10 -> 25600
  OFF_WG1 = 25600,    // 153x64  -> 160x64  : nk=5, nn=4  -> 10240
  OFF_WG2 = 35840,    // 64x32              : nk=2, nn=2  -> 2048
  OFF_WG3 = 37888,    // 32x8   -> 32x16    : nk=1, nn=1  -> 512
  OFF_WE1 = 38400,    // 8x(153x128 -> 160x128): nk=5, nn=8 -> 163840
  OFF_WE2 = 202240,   // 8x(128x64)            : nk=4, nn=4 -> 65536
  OFF_WE3 = 267776,   // 8x(64x8 -> 64x16)     : nk=2, nn=1 -> 8192
};                    // total 275968 halves = 539 KB of workspace

__global__ void pack_weights_kernel(const float* __restrict__ W,
                                    _Float16* __restrict__ out,
                                    int K, int N, int nk, int nn, int nmat) {
  int total = nmat * nk * nn * 512;
  for (int idx = blockIdx.x * blockDim.x + threadIdx.x; idx < total;
       idx += gridDim.x * blockDim.x) {
    int j    = idx & 15;
    int lane = (idx >> 4) & 31;
    int frag = idx >> 9;
    int kt   = frag % nk;
    int rem  = frag / nk;
    int nt   = rem % nn;
    int mat  = rem / nn;
    int n = nt * 16 + (lane & 15);
    int k = kt * 32 + ((lane >> 4) << 3) + (j & 7) + ((j >> 3) << 4);
    float v = (k < K && n < N) ? W[(size_t)mat * K * N + (size_t)k * N + n] : 0.0f;
    out[idx] = (_Float16)v;
  }
}

__device__ __forceinline__ v8f wmma16(v16h a, v16h b, v8f c) {
  return __builtin_amdgcn_wmma_f32_16x16x32_f16(false, a, false, b, (short)0, c,
                                                false, false);
}

// A fragment (16x32 f16, MxK) from an LDS tile with given row stride (halves).
__device__ __forceinline__ v16h a_frag(const _Float16* lds, int stride, int m,
                                       int kt, int h) {
  const _Float16* p = lds + m * stride + kt * 32 + h * 8;
  v8h lo = *(const v8h*)p;
  v8h hi = *(const v8h*)(p + 16);
  v16h r;
#pragma unroll
  for (int j = 0; j < 8; ++j) { r[j] = lo[j]; r[j + 8] = hi[j]; }
  return r;
}

// B fragment: one coalesced 32B load/lane from the pre-packed weight buffer.
__device__ __forceinline__ v16h b_frag(const _Float16* __restrict__ base,
                                       int frag, int lane) {
  return *(const v16h*)(base + ((size_t)frag << 9) + ((size_t)lane << 4));
}

__global__ __launch_bounds__(128) void moe_fused_kernel(
    const float* __restrict__ x,
    const float* __restrict__ ln_in_g, const float* __restrict__ ln_in_b,
    const float* __restrict__ bp,
    const float* __restrict__ bg1, const float* __restrict__ bg2,
    const float* __restrict__ bg3,
    const float* __restrict__ be1, const float* __restrict__ be2,
    const float* __restrict__ be3,
    const float* __restrict__ ln_out_g, const float* __restrict__ ln_out_b,
    const _Float16* __restrict__ wpk,
    float* __restrict__ out, float* __restrict__ gate_out) {
  // wave-private LDS (no cross-wave sharing -> no barriers; DS ops of a wave
  // are executed in order per the CDNA5 ISA).
  __shared__ __align__(16) _Float16 s_bufA[WAVES_PER_BLOCK][3072]; // xnorm / g1,g2 / h1,h2
  __shared__ __align__(16) _Float16 s_xp  [WAVES_PER_BLOCK][2560]; // x_proj [16][160]
  __shared__ __align__(16) float    s_lg  [WAVES_PER_BLOCK][256];  // logits [16][16]
  __shared__ __align__(16) float    s_gate[WAVES_PER_BLOCK][128];  // gating [16][8]
  __shared__ __align__(16) float    s_res [WAVES_PER_BLOCK][128];  // output [16][8]

  const int lane = threadIdx.x & 31;
  const int wave = threadIdx.x >> 5;
  const int n16  = lane & 15;
  const int h    = lane >> 4;
  const int tok0 = (blockIdx.x * WAVES_PER_BLOCK + wave) * TOKENS_PER_WAVE;

  _Float16* bufA = s_bufA[wave];
  _Float16* xp   = s_xp[wave];
  float*    lg   = s_lg[wave];
  float*    gate = s_gate[wave];
  float*    resb = s_res[wave];

  // ---- input LayerNorm over D_IN=153 -> f16 xnorm[16][160] (zero padded) ----
  {
    const float* xrow = x + (size_t)(tok0 + n16) * 153;
    const int k0 = h * 80;
    float s = 0.f, q = 0.f;
#pragma unroll 4
    for (int i = 0; i < 80; ++i) {
      int k = k0 + i;
      float v = (k < 153) ? xrow[k] : 0.0f;
      s += v; q += v * v;
    }
    s += __shfl_xor(s, 16);
    q += __shfl_xor(q, 16);
    const float mean = s * (1.0f / 153.0f);
    const float var  = q * (1.0f / 153.0f) - mean * mean;
    const float rstd = rsqrtf(var + 1e-5f);
#pragma unroll 4
    for (int i = 0; i < 80; ++i) {
      int k = k0 + i;
      float v = 0.0f;
      if (k < 153) v = (xrow[k] - mean) * rstd * ln_in_g[k] + ln_in_b[k];
      bufA[n16 * 160 + k] = (_Float16)v;
    }
  }

  // ---- x_proj = relu(xnorm @ Wp + bp) + xnorm -> f16 xp[16][160] ----
  for (int nt = 0; nt < 10; ++nt) {
    v8f acc = {0, 0, 0, 0, 0, 0, 0, 0};
#pragma unroll
    for (int kt = 0; kt < 5; ++kt)
      acc = wmma16(a_frag(bufA, 160, n16, kt, h),
                   b_frag(wpk + OFF_WP, nt * 5 + kt, lane), acc);
    const int n = nt * 16 + n16;
    const float bias = (n < 153) ? bp[n] : 0.0f;
#pragma unroll
    for (int v = 0; v < 8; ++v) {
      const int m = v + 8 * h;
      float r = fmaxf(acc[v] + bias, 0.0f) + (float)bufA[m * 160 + n];
      xp[m * 160 + n] = (_Float16)r;
    }
  }

  // ---- gating MLP + softmax -> gate[16][8] ----
  {
    _Float16* g1 = bufA;         // [16][64]
    _Float16* g2 = bufA + 1024;  // [16][32]
    for (int nt = 0; nt < 4; ++nt) {
      v8f acc = {0, 0, 0, 0, 0, 0, 0, 0};
#pragma unroll
      for (int kt = 0; kt < 5; ++kt)
        acc = wmma16(a_frag(xp, 160, n16, kt, h),
                     b_frag(wpk + OFF_WG1, nt * 5 + kt, lane), acc);
      const int n = nt * 16 + n16;
      const float bias = bg1[n];
#pragma unroll
      for (int v = 0; v < 8; ++v)
        g1[(v + 8 * h) * 64 + n] = (_Float16)fmaxf(acc[v] + bias, 0.0f);
    }
    for (int nt = 0; nt < 2; ++nt) {
      v8f acc = {0, 0, 0, 0, 0, 0, 0, 0};
#pragma unroll
      for (int kt = 0; kt < 2; ++kt)
        acc = wmma16(a_frag(g1, 64, n16, kt, h),
                     b_frag(wpk + OFF_WG2, nt * 2 + kt, lane), acc);
      const int n = nt * 16 + n16;
      const float bias = bg2[n];
#pragma unroll
      for (int v = 0; v < 8; ++v)
        g2[(v + 8 * h) * 32 + n] = (_Float16)fmaxf(acc[v] + bias, 0.0f);
    }
    {
      v8f acc = {0, 0, 0, 0, 0, 0, 0, 0};
      acc = wmma16(a_frag(g2, 32, n16, 0, h), b_frag(wpk + OFF_WG3, 0, lane), acc);
      const float bias = (n16 < 8) ? bg3[n16] : 0.0f;
#pragma unroll
      for (int v = 0; v < 8; ++v)
        lg[(v + 8 * h) * 16 + n16] = acc[v] + bias;
    }
    if (lane < 16) {  // one lane per token row: softmax over E=8
      float m = -3.4e38f;
#pragma unroll
      for (int e = 0; e < 8; ++e) m = fmaxf(m, lg[lane * 16 + e]);
      float ex[8];
      float ssum = 0.f;
#pragma unroll
      for (int e = 0; e < 8; ++e) { ex[e] = __expf(lg[lane * 16 + e] - m); ssum += ex[e]; }
      const float inv = 1.0f / ssum;
#pragma unroll
      for (int e = 0; e < 8; ++e) gate[lane * 8 + e] = ex[e] * inv;
    }
  }

  // ---- all 8 experts, gated accumulate into registers ----
  float oacc[8];
#pragma unroll
  for (int v = 0; v < 8; ++v) oacc[v] = 0.0f;
  _Float16* h1b = bufA;         // [16][128]
  _Float16* h2b = bufA + 2048;  // [16][64]
  for (int e = 0; e < 8; ++e) {
    for (int nt = 0; nt < 8; ++nt) {
      v8f acc = {0, 0, 0, 0, 0, 0, 0, 0};
#pragma unroll
      for (int kt = 0; kt < 5; ++kt)
        acc = wmma16(a_frag(xp, 160, n16, kt, h),
                     b_frag(wpk + OFF_WE1, (e * 8 + nt) * 5 + kt, lane), acc);
      const int n = nt * 16 + n16;
      const float bias = be1[e * 128 + n];
#pragma unroll
      for (int v = 0; v < 8; ++v)
        h1b[(v + 8 * h) * 128 + n] = (_Float16)fmaxf(acc[v] + bias, 0.0f);
    }
    for (int nt = 0; nt < 4; ++nt) {
      v8f acc = {0, 0, 0, 0, 0, 0, 0, 0};
#pragma unroll
      for (int kt = 0; kt < 4; ++kt)
        acc = wmma16(a_frag(h1b, 128, n16, kt, h),
                     b_frag(wpk + OFF_WE2, (e * 4 + nt) * 4 + kt, lane), acc);
      const int n = nt * 16 + n16;
      const float bias = be2[e * 64 + n];
#pragma unroll
      for (int v = 0; v < 8; ++v)
        h2b[(v + 8 * h) * 64 + n] = (_Float16)fmaxf(acc[v] + bias, 0.0f);
    }
    {
      v8f acc = {0, 0, 0, 0, 0, 0, 0, 0};
#pragma unroll
      for (int kt = 0; kt < 2; ++kt)
        acc = wmma16(a_frag(h2b, 64, n16, kt, h),
                     b_frag(wpk + OFF_WE3, e * 2 + kt, lane), acc);
      const float bias = (n16 < 8) ? be3[e * 8 + n16] : 0.0f;
#pragma unroll
      for (int v = 0; v < 8; ++v)
        oacc[v] += gate[(v + 8 * h) * 8 + e] * (acc[v] + bias);
    }
  }

  // ---- output LayerNorm over D_OUT=8 (padded cols are exactly 0) ----
  const float go = (n16 < 8) ? ln_out_g[n16] : 0.0f;
  const float bo = (n16 < 8) ? ln_out_b[n16] : 0.0f;
#pragma unroll
  for (int v = 0; v < 8; ++v) {
    float xv = oacc[v];
    float s = xv, q = xv * xv;
    s += __shfl_xor(s, 1); q += __shfl_xor(q, 1);
    s += __shfl_xor(s, 2); q += __shfl_xor(q, 2);
    s += __shfl_xor(s, 4); q += __shfl_xor(q, 4);
    const float mean = s * 0.125f;
    const float var  = q * 0.125f - mean * mean;
    const float rstd = rsqrtf(var + 1e-5f);
    if (n16 < 8)
      resb[(v + 8 * h) * 8 + n16] = (xv - mean) * rstd * go + bo;
  }

  // ---- coalesced stores: 16 tokens x 8 f32 = one float4 per lane ----
  {
    const float4 o  = *(const float4*)&resb[lane * 4];
    *(float4*)(out + (size_t)tok0 * 8 + lane * 4) = o;
    const float4 g4 = *(const float4*)&gate[lane * 4];
    *(float4*)(gate_out + (size_t)tok0 * 8 + lane * 4) = g4;
  }
}

extern "C" void kernel_launch(void* const* d_in, const int* in_sizes, int n_in,
                              void* d_out, int out_size, void* d_ws, size_t ws_size,
                              hipStream_t stream) {
  const float* x    = (const float*)d_in[0];
  const float* lig  = (const float*)d_in[1];
  const float* lib  = (const float*)d_in[2];
  const float* Wp   = (const float*)d_in[3];
  const float* bp   = (const float*)d_in[4];
  const float* Wg1  = (const float*)d_in[5];
  const float* bg1  = (const float*)d_in[6];
  const float* Wg2  = (const float*)d_in[7];
  const float* bg2  = (const float*)d_in[8];
  const float* Wg3  = (const float*)d_in[9];
  const float* bg3  = (const float*)d_in[10];
  const float* We1  = (const float*)d_in[11];
  const float* be1  = (const float*)d_in[12];
  const float* We2  = (const float*)d_in[13];
  const float* be2  = (const float*)d_in[14];
  const float* We3  = (const float*)d_in[15];
  const float* be3  = (const float*)d_in[16];
  const float* log_ = (const float*)d_in[17];
  const float* lob  = (const float*)d_in[18];

  _Float16* wpk = (_Float16*)d_ws;
  const int B = in_sizes[0] / 153;

  auto pack = [&](const float* W, int off, int K, int N, int nk, int nn, int nmat) {
    int total  = nmat * nk * nn * 512;
    int blocks = (total + 255) / 256;
    pack_weights_kernel<<<dim3(blocks), dim3(256), 0, stream>>>(
        W, wpk + off, K, N, nk, nn, nmat);
  };
  pack(Wp,  OFF_WP,  153, 153, 5, 10, 1);
  pack(Wg1, OFF_WG1, 153, 64,  5, 4,  1);
  pack(Wg2, OFF_WG2, 64,  32,  2, 2,  1);
  pack(Wg3, OFF_WG3, 32,  8,   1, 1,  1);
  pack(We1, OFF_WE1, 153, 128, 5, 8,  8);
  pack(We2, OFF_WE2, 128, 64,  4, 4,  8);
  pack(We3, OFF_WE3, 64,  8,   2, 1,  8);

  float* outp = (float*)d_out;
  float* gout = outp + (size_t)B * 8;
  const int blocks = B / TOKENS_PER_BLOCK;
  moe_fused_kernel<<<dim3(blocks), dim3(128), 0, stream>>>(
      x, lig, lib, bp, bg1, bg2, bg3, be1, be2, be3, log_, lob, wpk, outp, gout);
}
